// GQA_33801392619698
// MI455X (gfx1250) — compile-verified, matrix operand order checked by reading, not run on hardware
//
#include <hip/hip_runtime.h>
#include <hip/hip_bf16.h>

// ---------------------------------------------------------------------------
// GQA block for MI455X (gfx1250, wave32, WMMA).
// Pipeline: pack x/W to bf16 (once) -> [x@Wq, x@Wk, x@Wv] (bf16 WMMA GEMM)
//           -> RoPE+pack -> flash attention (bf16 WMMA, online softmax, bf16
//           out) -> attn@Wo (WMMA GEMM).
// All matmuls use v_wmma_f32_16x16x32_bf16 (f32 accumulation).
// GEMM staging is pure b128 copies (no conversion in the hot loop); all
// fragment loads are two ds/global b128 per fragment.
// ---------------------------------------------------------------------------

typedef __attribute__((ext_vector_type(16))) __bf16          v16bf;
typedef __attribute__((ext_vector_type(8)))  float           v8f;
typedef __attribute__((ext_vector_type(4)))  unsigned        v4u;
typedef __attribute__((ext_vector_type(8)))  unsigned        v8u;

constexpr int Bb   = 2;
constexpr int Ss   = 2048;
constexpr int Dd   = 2048;
constexpr int Hh   = 16;
constexpr int HKVv = 4;
constexpr int HDd  = 128;

__device__ __forceinline__ unsigned short f2bf(float f) {
  __bf16 b = (__bf16)f;                     // native cvt, RNE
  return __builtin_bit_cast(unsigned short, b);
}

__device__ __forceinline__ v8f zero8() {
  v8f z = {0.f, 0.f, 0.f, 0.f, 0.f, 0.f, 0.f, 0.f};
  return z;
}

__device__ __forceinline__ v8f wmma_bf16(v16bf a, v16bf b, v8f c) {
  // 8 args: (neg_a, A, neg_b, B, c_mod, C, reuse_a, reuse_b)
  return __builtin_amdgcn_wmma_f32_16x16x32_bf16(false, a, false, b, (short)0, c,
                                                 false, false);
}

// Fragment loader (A-pattern): 16x32 bf16 tile, element [m][k], from packed
// bf16-pair (uint) storage, row-major with `ldw` uints per row (multiple of 4
// => rows stay 16B aligned). Per ISA 7.12.2 16-bit layout:
//   lanes 0-15 : M=lane,     K in {0..7, 16..23}
//   lanes 16-31: M=lane-16,  K in {8..15, 24..31}
// Each lane performs exactly two 16B vector loads (ds_load_b128 /
// global_load_b128). Also used for B-fragments when the tile is stored
// transposed (row index = n, k contiguous).
__device__ __forceinline__ v16bf load_frag(const unsigned* src, int ldw,
                                           int row_off, int kw_off) {
  const int lane = threadIdx.x & 31;
  const int g = lane >> 4, m = lane & 15;
  const unsigned* row = src + (size_t)(row_off + m) * ldw + kw_off;
  const v4u lo = *(const v4u*)(row + 4 * g);       // k = 8g .. 8g+7
  const v4u hi = *(const v4u*)(row + 8 + 4 * g);   // k = 16+8g .. 16+8g+7
  v8u r;
  r[0] = lo[0]; r[1] = lo[1]; r[2] = lo[2]; r[3] = lo[3];
  r[4] = hi[0]; r[5] = hi[1]; r[6] = hi[2]; r[7] = hi[3];
  return __builtin_bit_cast(v16bf, r);
}

// ---------------------------------------------------------------------------
// One-time packing kernels (bandwidth-trivial, L2-resident).
// ---------------------------------------------------------------------------
__global__ void __launch_bounds__(256)
convert_bf16(const float* __restrict__ src, unsigned short* __restrict__ dst,
             size_t count2) {  // count2 = count/2; one packed pair per thread
  const size_t i = (size_t)blockIdx.x * blockDim.x + threadIdx.x;
  if (i >= count2) return;
  unsigned p = (unsigned)f2bf(src[2 * i]) | ((unsigned)f2bf(src[2 * i + 1]) << 16);
  ((unsigned*)dst)[i] = p;
}

// WT[n][k] = bf16(W[k][n]) : produces the exact layout the GEMM B-tile wants.
__global__ void __launch_bounds__(256)
transpose_to_bf16(const float* __restrict__ W, unsigned short* __restrict__ WT,
                  int K, int N) {
  const size_t idx = (size_t)blockIdx.x * blockDim.x + threadIdx.x;
  if (idx >= (size_t)K * N) return;
  const int k  = (int)(idx % K);
  const int nn = (int)(idx / K);
  WT[(size_t)nn * K + k] = f2bf(W[(size_t)k * N + nn]);
}

// ---------------------------------------------------------------------------
// GEMM: C(MxN, f32) = A(MxK, bf16 row-major) * BT(NxK, bf16 "col-major").
// Block tile 256x128x64, 256 threads (8 waves, 4x2 grid, 64x64 per wave).
// Staging: pure 16B copies, zero VALU. Per K-slice: 8 fragment loads feed
// 16 WMMAs (~1 b128 per WMMA); 32 WMMAs per wave between barriers.
// ---------------------------------------------------------------------------
#define BMt 256
#define BNt 128
#define BKt 64
#define AW  36   // uints per LDS row: 32 data + 4 pad (keeps 16B alignment)

__global__ void __launch_bounds__(256)
gemm_bf16_wmma(const unsigned short* __restrict__ A16,
               const unsigned short* __restrict__ BT16,
               float* __restrict__ C, int M, int N, int K) {
  __shared__ unsigned As[BMt * AW];    // [row m][k pairs]          (36 KB)
  __shared__ unsigned BsT[BNt * AW];   // [col n][k pairs]          (18 KB)

  const int tid  = threadIdx.x;
  const int wid  = tid >> 5;
  const int wm   = wid >> 1;                 // 0..3
  const int wn   = wid & 1;                  // 0..1
  const int bm0  = blockIdx.y * BMt;
  const int bn0  = blockIdx.x * BNt;
  const int lane = tid & 31, g = lane >> 4, n = lane & 15;

  v8f acc[4][4];
#pragma unroll
  for (int i = 0; i < 4; ++i)
#pragma unroll
    for (int j = 0; j < 4; ++j) acc[i][j] = zero8();

  const int bcol  = tid & 127;        // B column handled by this thread
  const int bhalf = tid >> 7;         // which 16-uint half of the B row

  for (int k0 = 0; k0 < K; k0 += BKt) {
    // A: one full 64-elem row slice per thread (32 uints, 16B aligned)
    const unsigned* ap = (const unsigned*)(A16 + (size_t)(bm0 + tid) * K + k0);
    // B: contiguous k-slice of one column of BT (16 uints per thread)
    const unsigned* bp =
        (const unsigned*)(BT16 + (size_t)(bn0 + bcol) * K + k0) + bhalf * 16;
    if (k0 + BKt < K) {                       // -> global_prefetch_b8
      __builtin_prefetch(ap + 32, 0, 0);
      __builtin_prefetch(bp + 32, 0, 0);
    }
#pragma unroll
    for (int i = 0; i < 8; ++i)
      *(v4u*)&As[tid * AW + 4 * i] = *(const v4u*)(ap + 4 * i);
#pragma unroll
    for (int i = 0; i < 4; ++i)
      *(v4u*)&BsT[bcol * AW + bhalf * 16 + 4 * i] = *(const v4u*)(bp + 4 * i);
    __syncthreads();

#pragma unroll
    for (int ks = 0; ks < 2; ++ks) {
      v16bf af[4];
#pragma unroll
      for (int mi = 0; mi < 4; ++mi)
        af[mi] = load_frag(As, AW, wm * 64 + mi * 16, ks * 16);
#pragma unroll
      for (int nj = 0; nj < 4; ++nj) {
        v16bf bfj = load_frag(BsT, AW, wn * 64 + nj * 16, ks * 16);
#pragma unroll
        for (int mi = 0; mi < 4; ++mi)
          acc[mi][nj] = wmma_bf16(af[mi], bfj, acc[mi][nj]);
      }
    }
    __syncthreads();
  }

#pragma unroll
  for (int mi = 0; mi < 4; ++mi)
#pragma unroll
    for (int nj = 0; nj < 4; ++nj)
#pragma unroll
      for (int r = 0; r < 8; ++r) {
        const int row = bm0 + wm * 64 + mi * 16 + r + 8 * g;
        const int col = bn0 + wn * 64 + nj * 16 + n;
        C[(size_t)row * N + col] = acc[mi][nj][r];
      }
}

// ---------------------------------------------------------------------------
// RoPE (concat-style, per reference) + transpose to (b, head, s, hd) bf16.
// ---------------------------------------------------------------------------
__global__ void __launch_bounds__(256)
rope_pack(const float* __restrict__ src, unsigned short* __restrict__ dst,
          int nheads) {
  const size_t idx = (size_t)blockIdx.x * blockDim.x + threadIdx.x;
  const size_t total = (size_t)Bb * nheads * Ss * (HDd / 2);
  if (idx >= total) return;
  const int j    = (int)(idx % (HDd / 2));
  const int s    = (int)((idx / (HDd / 2)) % Ss);
  const int head = (int)((idx / ((size_t)(HDd / 2) * Ss)) % nheads);
  const int b    = (int)(idx / ((size_t)(HDd / 2) * Ss * nheads));

  const float* p = src + ((size_t)b * Ss + s) * ((size_t)nheads * HDd) + head * HDd;
  const float x1 = p[2 * j], x2 = p[2 * j + 1];
  const float theta = __powf(10000.0f, -(float)j / 64.0f);
  const float ang = (float)s * theta;
  float sv, cv;
  __sincosf(ang, &sv, &cv);

  unsigned short* q = dst + (((size_t)b * nheads + head) * Ss + s) * HDd;
  q[j]      = f2bf(x1 * cv - x2 * sv);
  q[j + 64] = f2bf(x1 * sv + x2 * cv);
}

// V packed TRANSPOSED: (b, kvh, hd, s) bf16, so PV B-fragments (fixed hd,
// contiguous keys) become two 16B global loads per fragment.
__global__ void __launch_bounds__(256)
pack_v(const float* __restrict__ src, unsigned short* __restrict__ dst) {
  const size_t idx = (size_t)blockIdx.x * blockDim.x + threadIdx.x;
  const size_t total = (size_t)Bb * HKVv * Ss * HDd;
  if (idx >= total) return;
  const int hd   = (int)(idx % HDd);
  const int s    = (int)((idx / HDd) % Ss);
  const int head = (int)((idx / ((size_t)HDd * Ss)) % HKVv);
  const int b    = (int)(idx / ((size_t)HDd * Ss * HKVv));
  dst[(((size_t)b * HKVv + head) * HDd + hd) * Ss + s] =
      f2bf(src[((size_t)b * Ss + s) * (HKVv * HDd) + head * HDd + hd]);
}

// ---------------------------------------------------------------------------
// Causal flash attention. One wave (32 threads) per (b, h, 32-query tile:
// two 16-row subtiles sharing every K/V fragment). Keys in 32-wide chunks:
// 16 WMMAs QK^T + 16 WMMAs PV per chunk, online softmax with shfl_xor
// row-reductions (C-layout stripes a row over 16 lanes), P staged through
// LDS into A-fragment layout, V consumed transposed (hd-major).
// Output written directly as bf16 row-major (b, s, H*HD) = A of final GEMM.
// ---------------------------------------------------------------------------
__global__ void __launch_bounds__(32)
flash_attn(const unsigned short* __restrict__ Qb,
           const unsigned short* __restrict__ Kb,
           const unsigned short* __restrict__ Vt,
           unsigned short* __restrict__ Ob) {
  const int q0  = blockIdx.x * 32;
  const int h   = blockIdx.y;
  const int b   = blockIdx.z;
  const int kvh = h / (Hh / HKVv);
  const unsigned* Qw  = (const unsigned*)(Qb + (((size_t)b * Hh   + h)   * Ss) * HDd);
  const unsigned* Kw  = (const unsigned*)(Kb + (((size_t)b * HKVv + kvh) * Ss) * HDd);
  const unsigned* Vtw = (const unsigned*)(Vt + (((size_t)b * HKVv + kvh) * HDd) * Ss);
  const int QKW = HDd / 2;   // uints per Q/K row
  const int VW  = Ss / 2;    // uints per V^T row

  const int lane = threadIdx.x, g = lane >> 4, n = lane & 15;
  const float scale = 0.08838834764831845f;  // 1/sqrt(128)

  v16bf qa[2][4];
#pragma unroll
  for (int qm = 0; qm < 2; ++qm)
#pragma unroll
    for (int t = 0; t < 4; ++t)
      qa[qm][t] = load_frag(Qw, QKW, q0 + qm * 16, 16 * t);

  v8f acc[2][8];
  float m_i[2][8], l_i[2][8];
#pragma unroll
  for (int qm = 0; qm < 2; ++qm) {
#pragma unroll
    for (int t = 0; t < 8; ++t) acc[qm][t] = zero8();
#pragma unroll
    for (int r = 0; r < 8; ++r) { m_i[qm][r] = -1e30f; l_i[qm][r] = 0.f; }
  }

  __shared__ unsigned short Pshm[2][16 * 32];  // two 16x32 P tiles, ld=32

  for (int k0 = 0; k0 < q0 + 32; k0 += 32) {
    v8f s0[2], s1[2];
#pragma unroll
    for (int qm = 0; qm < 2; ++qm) { s0[qm] = zero8(); s1[qm] = zero8(); }
#pragma unroll
    for (int t = 0; t < 4; ++t) {
      // B[k][n] = K[k0+n][hd]: transposed access == A-fragment pattern
      v16bf kb0 = load_frag(Kw, QKW, k0,      16 * t);
      v16bf kb1 = load_frag(Kw, QKW, k0 + 16, 16 * t);
#pragma unroll
      for (int qm = 0; qm < 2; ++qm) {
        s0[qm] = wmma_bf16(qa[qm][t], kb0, s0[qm]);
        s1[qm] = wmma_bf16(qa[qm][t], kb1, s1[qm]);
      }
    }

#pragma unroll
    for (int qm = 0; qm < 2; ++qm) {
      float p0[8], p1[8], alpha[8];
#pragma unroll
      for (int r = 0; r < 8; ++r) {
        const int row = q0 + qm * 16 + r + 8 * g;
        float v0 = (k0 + n      <= row) ? s0[qm][r] * scale : -1e30f;
        float v1 = (k0 + 16 + n <= row) ? s1[qm][r] * scale : -1e30f;
        float mloc = fmaxf(v0, v1);
#pragma unroll
        for (int off = 1; off < 16; off <<= 1)
          mloc = fmaxf(mloc, __shfl_xor(mloc, off, 32));
        const float mnew = fmaxf(m_i[qm][r], mloc);
        alpha[r] = __expf(m_i[qm][r] - mnew);
        m_i[qm][r] = mnew;
        p0[r] = __expf(v0 - mnew);
        p1[r] = __expf(v1 - mnew);
        float ls = p0[r] + p1[r];
#pragma unroll
        for (int off = 1; off < 16; off <<= 1) ls += __shfl_xor(ls, off, 32);
        l_i[qm][r] = l_i[qm][r] * alpha[r] + ls;
      }
#pragma unroll
      for (int t = 0; t < 8; ++t)
#pragma unroll
        for (int r = 0; r < 8; ++r) acc[qm][t][r] *= alpha[r];
      // Stage P (C-layout regs) -> LDS row-major -> A-fragment layout.
#pragma unroll
      for (int r = 0; r < 8; ++r) {
        Pshm[qm][(r + 8 * g) * 32 + n]      = f2bf(p0[r]);
        Pshm[qm][(r + 8 * g) * 32 + 16 + n] = f2bf(p1[r]);
      }
    }

    v16bf pa0 = load_frag((const unsigned*)&Pshm[0][0], 16, 0, 0);
    v16bf pa1 = load_frag((const unsigned*)&Pshm[1][0], 16, 0, 0);
#pragma unroll
    for (int t = 0; t < 8; ++t) {
      // B[k][n] = V[k0+k][n0+n] = Vt[n0+n][k0+k]: A-fragment pattern on V^T
      v16bf vb = load_frag(Vtw, VW, 16 * t, k0 >> 1);
      acc[0][t] = wmma_bf16(pa0, vb, acc[0][t]);
      acc[1][t] = wmma_bf16(pa1, vb, acc[1][t]);
    }
  }

  // Write attention output as bf16 (b, s, H*HD): A-matrix of final GEMM.
#pragma unroll
  for (int qm = 0; qm < 2; ++qm)
#pragma unroll
    for (int t = 0; t < 8; ++t)
#pragma unroll
      for (int r = 0; r < 8; ++r) {
        const int row = q0 + qm * 16 + r + 8 * g;
        const float invl = 1.0f / l_i[qm][r];
        Ob[((size_t)b * Ss + row) * Dd + h * HDd + 16 * t + n] =
            f2bf(acc[qm][t][r] * invl);
      }
}

// ---------------------------------------------------------------------------
// Launch
// ---------------------------------------------------------------------------
extern "C" void kernel_launch(void* const* d_in, const int* in_sizes, int n_in,
                              void* d_out, int out_size, void* d_ws,
                              size_t ws_size, hipStream_t stream) {
  const float* x  = (const float*)d_in[0];  // (B,S,D)
  const float* Wq = (const float*)d_in[1];  // (D, H*HD)
  const float* Wk = (const float*)d_in[2];  // (D, HKV*HD)
  const float* Wv = (const float*)d_in[3];  // (D, HKV*HD)
  const float* Wo = (const float*)d_in[4];  // (H*HD, D)
  // d_in[5] = causal mask, not needed (mask applied analytically)
  float* out = (float*)d_out;

  const size_t M = (size_t)Bb * Ss;  // 4096
  // Workspace layout (~108 MB total)
  char* ws = (char*)d_ws;
  float* Qraw = (float*)ws;                                   // M*2048 f32 (32MB)
  float* Kraw = Qraw + M * (size_t)(Hh * HDd);                // M*512  f32
  float* Vraw = Kraw + M * (size_t)(HKVv * HDd);              // M*512  f32
  unsigned short* Qb  = (unsigned short*)(Vraw + M * (size_t)(HKVv * HDd));
  unsigned short* Kb  = Qb  + (size_t)Bb * Hh * Ss * HDd;
  unsigned short* Vt  = Kb  + (size_t)Bb * HKVv * Ss * HDd;
  unsigned short* xb  = Vt  + (size_t)Bb * HKVv * Ss * HDd;   // M*2048 bf16
  unsigned short* WqT = xb  + M * (size_t)Dd;                 // 2048x2048 bf16
  unsigned short* WkT = WqT + (size_t)Dd * (Hh * HDd);        // 512x2048 bf16
  unsigned short* WvT = WkT + (size_t)Dd * (HKVv * HDd);      // 512x2048 bf16
  unsigned short* WoT = WvT + (size_t)Dd * (HKVv * HDd);      // 2048x2048 bf16
  // Flash output reuses the Qraw region (RoPE consumed it earlier in-stream).
  unsigned short* Ob  = (unsigned short*)Qraw;                // M*2048 bf16

  // 0) One-time bf16 packing of activations and (transposed) weights
  {
    const size_t nx2 = M * (size_t)Dd / 2;
    convert_bf16<<<(int)((nx2 + 255) / 256), 256, 0, stream>>>(x, xb, nx2);
    const size_t nq = (size_t)Dd * (Hh * HDd);
    transpose_to_bf16<<<(int)((nq + 255) / 256), 256, 0, stream>>>(Wq, WqT, Dd, Hh * HDd);
    const size_t nk = (size_t)Dd * (HKVv * HDd);
    transpose_to_bf16<<<(int)((nk + 255) / 256), 256, 0, stream>>>(Wk, WkT, Dd, HKVv * HDd);
    transpose_to_bf16<<<(int)((nk + 255) / 256), 256, 0, stream>>>(Wv, WvT, Dd, HKVv * HDd);
    const size_t no = (size_t)(Hh * HDd) * Dd;
    transpose_to_bf16<<<(int)((no + 255) / 256), 256, 0, stream>>>(Wo, WoT, Hh * HDd, Dd);
  }

  // 1) QKV projections (bf16 WMMA GEMMs, pure-copy staging)
  gemm_bf16_wmma<<<dim3(Dd / BNt, (int)(M / BMt)), 256, 0, stream>>>(
      xb, WqT, Qraw, (int)M, Dd, Dd);
  gemm_bf16_wmma<<<dim3((HKVv * HDd) / BNt, (int)(M / BMt)), 256, 0, stream>>>(
      xb, WkT, Kraw, (int)M, HKVv * HDd, Dd);
  gemm_bf16_wmma<<<dim3((HKVv * HDd) / BNt, (int)(M / BMt)), 256, 0, stream>>>(
      xb, WvT, Vraw, (int)M, HKVv * HDd, Dd);

  // 2) RoPE + transpose/pack to bf16 (V transposed to hd-major)
  {
    const size_t nq = (size_t)Bb * Hh * Ss * (HDd / 2);
    rope_pack<<<(int)((nq + 255) / 256), 256, 0, stream>>>(Qraw, Qb, Hh);
    const size_t nk = (size_t)Bb * HKVv * Ss * (HDd / 2);
    rope_pack<<<(int)((nk + 255) / 256), 256, 0, stream>>>(Kraw, Kb, HKVv);
    const size_t nv = (size_t)Bb * HKVv * Ss * HDd;
    pack_v<<<(int)((nv + 255) / 256), 256, 0, stream>>>(Vraw, Vt);
  }

  // 3) Causal flash attention (32-query tiles, bf16 output)
  flash_attn<<<dim3(Ss / 32, Hh, Bb), 32, 0, stream>>>(Qb, Kb, Vt, Ob);

  // 4) Output projection
  gemm_bf16_wmma<<<dim3(Dd / BNt, (int)(M / BMt)), 256, 0, stream>>>(
      Ob, WoT, out, (int)M, Dd, Dd);
}
